// Router_69123203661942
// MI455X (gfx1250) — matrix operand order repaired and checked by reading, not compiled, not run on hardware
//
#include <hip/hip_runtime.h>
#include <hip/hip_bf16.h>
#include <math.h>

// ---------------- CDNA5 / gfx1250 WMMA types ----------------
typedef __attribute__((ext_vector_type(16))) __bf16 v16bf;
typedef __attribute__((ext_vector_type(8)))  __bf16 v8bf;
typedef __attribute__((ext_vector_type(4)))  __bf16 v4bf;
typedef __attribute__((ext_vector_type(8)))  float  v8f;
typedef __attribute__((ext_vector_type(4)))  float  v4f;

// Problem constants (reference: T=8192, D=4096, E=64, K=8)
#define EXPERTS      64
#define TOPK         8
#define KCHUNK       128                  // K-chunk staged in LDS
#define ROWB         (KCHUNK + 8)         // bf16/row (+16B pad -> conflict-free banks)
#define WAVES        4
#define NTHREADS     (WAVES * 32)
#define TOK_PER_BLK  64
#define XREGS        (TOK_PER_BLK * (KCHUNK / 4) / NTHREADS)   // 16 float4 per thread
// LDS: [Wg 64][Wc 64][x buf0 64][x buf1 64] rows, each ROWB bf16
#define LDS_BYTES    ((2u * EXPERTS + 2u * TOK_PER_BLK) * ROWB * 2u)   // 69,632 B

__global__ __launch_bounds__(NTHREADS, 1) void Router_69123203661942_kernel(
    const float* __restrict__ x,       // [T, D]
    const float* __restrict__ Wg,      // [E, D]
    const float* __restrict__ Wc,      // [E, D]
    const float* __restrict__ escale,  // [E]
    const float* __restrict__ ebias,   // [E]
    float* __restrict__ wout,          // [T, TOPK]
    int*   __restrict__ iout,          // [T, TOPK]
    int D)
{
    extern __shared__ __align__(16) char smem_raw[];
    __bf16* wlds = (__bf16*)smem_raw;                          // [2][E][ROWB]
    __bf16* xlds = wlds + (size_t)2 * EXPERTS * ROWB;          // [2][64][ROWB]
    float*  slds = (float*)smem_raw;                           // phase 2: [64][E]

    const int tid     = threadIdx.x;
    const int lane    = tid & 31;
    const int wave    = tid >> 5;
    const int l15     = lane & 15;
    const int halfSel = (lane >> 4) & 1;         // 0: lanes 0-15, 1: lanes 16-31
    const int mhalf   = wave & 1;                // which 32-token half
    const int nhalf   = wave >> 1;               // which 32-expert half
    const int tokBase = blockIdx.x * TOK_PER_BLK;

    v8f accG[2][2], accC[2][2];
#pragma unroll
    for (int mi = 0; mi < 2; ++mi)
#pragma unroll
        for (int ni = 0; ni < 2; ++ni)
#pragma unroll
            for (int j = 0; j < 8; ++j) { accG[mi][ni][j] = 0.0f; accC[mi][ni][j] = 0.0f; }

    const int nChunks = D / KCHUNK;

    // ---- prologue: issue x loads for chunk 0 (held in regs) ----
    v4f xs[XREGS];
#pragma unroll
    for (int j = 0; j < XREGS; ++j) {
        const int i  = tid + j * NTHREADS;
        const int tl = i >> 5;                  // KCHUNK/4 = 32 float4 per row
        const int c  = (i & 31) << 2;
        xs[j] = *(const v4f*)(x + (size_t)(tokBase + tl) * D + c);
    }

    int buf = 0;
    for (int ci = 0; ci < nChunks; ++ci) {
        const int kc = ci * KCHUNK;
        __bf16* xcur = xlds + (size_t)buf * TOK_PER_BLK * ROWB;

        // ---- commit pending x regs (chunk ci) -> current x buffer ----
#pragma unroll
        for (int j = 0; j < XREGS; ++j) {
            const int i  = tid + j * NTHREADS;
            const int tl = i >> 5;
            const int c  = (i & 31) << 2;
            v4bf bv;
#pragma unroll
            for (int t = 0; t < 4; ++t) bv[t] = (__bf16)xs[j][t];
            *(v4bf*)(xcur + (size_t)tl * ROWB + c) = bv;
        }
        // ---- issue x loads for chunk ci+1; they fly through the WMMA loop ----
        if (ci + 1 < nChunks) {
#pragma unroll
            for (int j = 0; j < XREGS; ++j) {
                const int i  = tid + j * NTHREADS;
                const int tl = i >> 5;
                const int c  = (i & 31) << 2;
                xs[j] = *(const v4f*)(x + (size_t)(tokBase + tl) * D + kc + KCHUNK + c);
            }
        }
        // ---- stage W chunk (L2-resident, short latency, immediate) ----
        for (int i = tid; i < EXPERTS * (KCHUNK / 4); i += NTHREADS) {
            const int e = i >> 5;
            const int c = (i & 31) << 2;
            v4f g = *(const v4f*)(Wg + (size_t)e * D + kc + c);
            v4f q = *(const v4f*)(Wc + (size_t)e * D + kc + c);
            v4bf bg, bq;
#pragma unroll
            for (int t = 0; t < 4; ++t) { bg[t] = (__bf16)g[t]; bq[t] = (__bf16)q[t]; }
            *(v4bf*)(wlds + (size_t)e * ROWB + c)             = bg;
            *(v4bf*)(wlds + (size_t)(EXPERTS + e) * ROWB + c) = bq;
        }
        __syncthreads();

        // ---- inner K loop: 4 steps of 32, pure LDS + WMMA ----
        for (int ks = 0; ks < KCHUNK; ks += 32) {
            // A-fragment LDS loads (2 frags, 4 x b128)
            v8bf a_lo[2], a_hi[2];
#pragma unroll
            for (int mi = 0; mi < 2; ++mi) {
                const __bf16* ap = xcur + (size_t)(mhalf * 32 + mi * 16 + l15) * ROWB
                                   + ks + (halfSel ? 16 : 0);
                a_lo[mi] = *(const v8bf*)(ap);
                a_hi[mi] = *(const v8bf*)(ap + 8);
            }
            // B-fragment LDS loads in WMMA-use order (4 frags, 8 x b128)
            v8bf bg_lo[2], bg_hi[2], bc_lo[2], bc_hi[2];
#pragma unroll
            for (int ni = 0; ni < 2; ++ni) {
                const int nt = nhalf * 2 + ni;
                const __bf16* bp = wlds + (size_t)(nt * 16 + l15) * ROWB + ks + (halfSel ? 16 : 0);
                bg_lo[ni] = *(const v8bf*)(bp);
                bg_hi[ni] = *(const v8bf*)(bp + 8);
                const __bf16* bp2 = bp + (size_t)EXPERTS * ROWB;
                bc_lo[ni] = *(const v8bf*)(bp2);
                bc_hi[ni] = *(const v8bf*)(bp2 + 8);
            }

            v16bf A[2], Bg[2], Bc[2];
#pragma unroll
            for (int mi = 0; mi < 2; ++mi)
#pragma unroll
                for (int t = 0; t < 8; ++t) { A[mi][t] = a_lo[mi][t]; A[mi][t + 8] = a_hi[mi][t]; }
#pragma unroll
            for (int ni = 0; ni < 2; ++ni)
#pragma unroll
                for (int t = 0; t < 8; ++t) {
                    Bg[ni][t] = bg_lo[ni][t]; Bg[ni][t + 8] = bg_hi[ni][t];
                    Bc[ni][t] = bc_lo[ni][t]; Bc[ni][t + 8] = bc_hi[ni][t];
                }

#pragma unroll
            for (int ni = 0; ni < 2; ++ni) {
#pragma unroll
                for (int mi = 0; mi < 2; ++mi)
                    accG[mi][ni] = __builtin_amdgcn_wmma_f32_16x16x32_bf16(
                        false, A[mi], false, Bg[ni], (short)0, accG[mi][ni], false, false);
#pragma unroll
                for (int mi = 0; mi < 2; ++mi)
                    accC[mi][ni] = __builtin_amdgcn_wmma_f32_16x16x32_bf16(
                        false, A[mi], false, Bc[ni], (short)0, accC[mi][ni], false, false);
            }
        }
        __syncthreads();   // WMMA reads done -> W buffer may be rewritten next iter
        buf ^= 1;
    }

    // ---- epilogue: scores = |cls * silu(gate)| -> LDS ----
    // C layout: VGPR j holds M=j (lanes 0-15) / M=j+8 (lanes 16-31), N = l15.
#pragma unroll
    for (int mi = 0; mi < 2; ++mi)
#pragma unroll
        for (int ni = 0; ni < 2; ++ni) {
            const int e = nhalf * 32 + ni * 16 + l15;
#pragma unroll
            for (int j = 0; j < 8; ++j) {
                float g = accG[mi][ni][j];
                float c = accC[mi][ni][j];
                float sil = g * __builtin_amdgcn_rcpf(1.0f + __expf(-g));
                float s = fabsf(c * sil);
                const int tl = mhalf * 32 + mi * 16 + j + (halfSel ? 8 : 0);
                slds[tl * EXPERTS + e] = s;
            }
        }
    __syncthreads();

    // ---- one thread per token: fp32 softmax + biased top-8 + gather ----
    if (tid < TOK_PER_BLK) {
        float* row = slds + tid * EXPERTS;
        float mx = -INFINITY;
#pragma unroll 8
        for (int e = 0; e < EXPERTS; ++e) mx = fmaxf(mx, row[e]);
        float sum = 0.0f;
#pragma unroll 8
        for (int e = 0; e < EXPERTS; ++e) { float p = __expf(row[e] - mx); row[e] = p; sum += p; }
        const float inv = __builtin_amdgcn_rcpf(sum);

        unsigned long long used = 0ull;
        const int t = tokBase + tid;
        for (int k = 0; k < TOPK; ++k) {
            float best = -INFINITY; int bi = 0;
            for (int e = 0; e < EXPERTS; ++e) {
                if ((used >> e) & 1ull) continue;
                float b = row[e] * inv + ebias[e];
                if (b > best) { best = b; bi = e; }
            }
            used |= (1ull << bi);
            float p = row[bi] * inv;
            wout[(size_t)t * TOPK + k] = 1.0f + p * escale[bi];
            iout[(size_t)t * TOPK + k] = bi;
        }
    }
}

extern "C" void kernel_launch(void* const* d_in, const int* in_sizes, int n_in,
                              void* d_out, int out_size, void* d_ws, size_t ws_size,
                              hipStream_t stream) {
    const float* x      = (const float*)d_in[0];
    const float* Wg     = (const float*)d_in[1];
    const float* Wc     = (const float*)d_in[2];
    const float* escale = (const float*)d_in[3];
    const float* ebias  = (const float*)d_in[4];

    const int E = in_sizes[3];            // 64
    const int D = in_sizes[1] / E;        // 4096
    const int T = in_sizes[0] / D;        // 8192

    float* wout = (float*)d_out;
    int*   iout = (int*)d_out + (size_t)T * TOPK;   // indices follow weights, flat

    dim3 grid(T / TOK_PER_BLK);           // 128 blocks
    dim3 block(NTHREADS);                 // 128 threads = 4 wave32
    Router_69123203661942_kernel<<<grid, block, LDS_BYTES, stream>>>(
        x, Wg, Wc, escale, ebias, wout, iout, D);
}
